// TFGPT2MainLayer_42064909697335
// MI455X (gfx1250) — matrix operand order repaired and checked by reading, not compiled, not run on hardware
//
#include <hip/hip_runtime.h>

// ---------------- types ----------------
typedef __attribute__((ext_vector_type(16))) __bf16 v16bf;
typedef __attribute__((ext_vector_type(8)))  __bf16 v8bf;
typedef __attribute__((ext_vector_type(8)))  float  v8f;
typedef int v4i_gcc __attribute__((vector_size(16)));   // pointee type expected by async-LDS builtin

#define TOK   4096   // B*S
#define DMODEL 768
#define NHEAD 12
#define HD    64
#define FDIM  3072
#define SEQ   512
#define NLAYER 12
#define D3    2304

__device__ __forceinline__ v8f vzero8() {
  v8f z = {0.f,0.f,0.f,0.f,0.f,0.f,0.f,0.f};
  return z;
}

__device__ __forceinline__ unsigned short f2bf(float f) {
  unsigned int u = __float_as_uint(f);
  unsigned int r = (u + 0x7FFFu + ((u >> 16) & 1u)) >> 16;
  return (unsigned short)r;
}

__device__ __forceinline__ v8f wmma_bf16(v16bf a, v16bf b, v8f c) {
  return __builtin_amdgcn_wmma_f32_16x16x32_bf16(false, a, false, b, (short)0, c, false, false);
}

__device__ __forceinline__ float gelu_tanh(float x) {
  float x3 = x * x * x;
  return 0.5f * x * (1.0f + tanhf(0.7978845608028654f * (x + 0.044715f * x3)));
}

// -------- async global->LDS copy (16B per lane), guarded for toolchain --------
#if defined(__has_builtin)
#if __has_builtin(__builtin_amdgcn_global_load_async_to_lds_b128)
#define HAVE_ASYNC_LDS 1
#endif
#if __has_builtin(__builtin_amdgcn_s_wait_asynccnt)
#define HAVE_WAIT_ASYNC 1
#endif
#endif

__device__ __forceinline__ void cp_async16(unsigned short* lds, const unsigned short* g) {
#ifdef HAVE_ASYNC_LDS
  __builtin_amdgcn_global_load_async_to_lds_b128(
      (__attribute__((address_space(1))) v4i_gcc*)g,
      (__attribute__((address_space(3))) v4i_gcc*)lds, 0, 0);
#else
  *reinterpret_cast<v8bf*>(lds) = *reinterpret_cast<const v8bf*>(g);
#endif
}

#ifdef HAVE_WAIT_ASYNC
#define WAIT_ASYNC(n) __builtin_amdgcn_s_wait_asynccnt(n)
#else
#define WAIT_ASYNC(n) asm volatile("s_wait_asynccnt %0" ::"i"(n) : "memory")
#endif

// ---------------- embeddings: x = wte[id] + wpe[pos] ----------------
__global__ __launch_bounds__(256) void k_embed(const int* __restrict__ ids,
                                               const float* __restrict__ wte,
                                               const float* __restrict__ wpe,
                                               float* __restrict__ x) {
  int i = blockIdx.x * 256 + threadIdx.x;            // over TOK*DMODEL
  int m = i / DMODEL;
  int d = i - m * DMODEL;
  int s = m & (SEQ - 1);
  x[i] = wte[(size_t)ids[m] * DMODEL + d] + wpe[(size_t)s * DMODEL + d];
}

// ---------------- pack f32 [K,N] weight -> bf16 WMMA B-fragment layout ----------------
// packed index: ((nt*(K/32)+kc)*32 + lane)*16 + i
//   lane: col = nt*16 + (lane&15), K = kc*32 + (lane>>4)*16 + i
__global__ __launch_bounds__(256) void k_pack(const float* __restrict__ w,
                                              unsigned short* __restrict__ out,
                                              int K, int N) {
  int i = blockIdx.x * 256 + threadIdx.x;            // over N*K/16 fragment-lanes
  int lane = i & 31;
  int kcn = K >> 5;
  int kc = (i >> 5) % kcn;
  int nt = (i >> 5) / kcn;
  int col = nt * 16 + (lane & 15);
  int kbase = kc * 32 + (lane >> 4) * 16;
  unsigned short* o = out + (size_t)i * 16;
#pragma unroll
  for (int e = 0; e < 16; ++e)
    o[e] = f2bf(w[(size_t)(kbase + e) * N + col]);
}

// ---------------- layernorm: one wave per row ----------------
template <bool F32OUT>
__global__ __launch_bounds__(256) void k_ln(const float* __restrict__ x,
                                            const float* __restrict__ g,
                                            const float* __restrict__ bta,
                                            unsigned short* __restrict__ ob,
                                            float* __restrict__ of) {
  int row = blockIdx.x * 8 + (threadIdx.x >> 5);
  int lane = threadIdx.x & 31;
  const float* r = x + (size_t)row * DMODEL;
  float s = 0.f;
  for (int j = lane; j < DMODEL; j += 32) s += r[j];
#pragma unroll
  for (int m = 16; m >= 1; m >>= 1) s += __shfl_xor(s, m, 32);
  float mean = s * (1.0f / DMODEL);
  float v = 0.f;
  for (int j = lane; j < DMODEL; j += 32) { float t = r[j] - mean; v += t * t; }
#pragma unroll
  for (int m = 16; m >= 1; m >>= 1) v += __shfl_xor(v, m, 32);
  float rstd = rsqrtf(v * (1.0f / DMODEL) + 1e-5f);
  for (int j = lane; j < DMODEL; j += 32) {
    float y = (r[j] - mean) * rstd * g[j] + bta[j];
    if (F32OUT) of[(size_t)row * DMODEL + j] = y;
    else        ob[(size_t)row * DMODEL + j] = f2bf(y);
  }
}

// ---------------- GEMM: C[M=4096, N] = A[M,K](bf16) @ Wpacked + bias ----------------
// Block tile 64x256: 8 waves (2 M x 4 N), each wave 32x64 (2x4 WMMA frags).
// K chunks of 32 staged in LDS with double-buffered async global->LDS copies.
// MODE 0: QKV split epilogue (q scaled 0.125, V transposed)
// MODE 1: residual add into f32 stream
// MODE 2: GELU -> bf16
template <int MODE>
__global__ __launch_bounds__(256) void k_gemm(const unsigned short* __restrict__ A,
                                              const unsigned short* __restrict__ Wp,
                                              const float* __restrict__ bias,
                                              int K, int N,
                                              float* __restrict__ xout,
                                              unsigned short* __restrict__ qb,
                                              unsigned short* __restrict__ kb,
                                              unsigned short* __restrict__ vtb,
                                              unsigned short* __restrict__ hout) {
  __shared__ __align__(64) unsigned short sA[2][64 * 32];     //  8 KB
  __shared__ __align__(64) unsigned short sB[2][16 * 512];    // 32 KB

  int tid = threadIdx.x;
  int widx = tid >> 5, lane = tid & 31;
  int lrow = lane & 15, half = lane >> 4;
  int wm = widx >> 2, wn = widx & 3;
  int bm = blockIdx.x & 63;            // M/64 == 64
  int bn = blockIdx.x >> 6;
  int kcn = K >> 5;
  size_t ntBase = (size_t)bn * 16;

  // copy roles: A panel 64 rows x 32 elems (4 x 16B segs per row)
  int arow = tid >> 2, aseg = tid & 3;
  const unsigned short* aG = A + (size_t)(bm * 64 + arow) * K + aseg * 8;
  // B panel: 16 n-tiles x 512 elems; each thread copies 4 x 16B
  int cElem[4];
  const unsigned short* bG[4];
#pragma unroll
  for (int i = 0; i < 4; ++i) {
    int c = (tid * 4 + i) * 8;
    int nt = c >> 9, off = c & 511;
    cElem[i] = c;
    bG[i] = Wp + ((ntBase + nt) * (size_t)kcn) * 512 + off;
  }

  v8f acc[2][4];
#pragma unroll
  for (int i = 0; i < 2; ++i)
#pragma unroll
    for (int j = 0; j < 4; ++j) acc[i][j] = vzero8();

  auto issue_chunk = [&](int kc, int b) {
    cp_async16(&sA[b][arow * 32 + aseg * 8], aG + (size_t)kc * 32);
#pragma unroll
    for (int i = 0; i < 4; ++i)
      cp_async16(&sB[b][cElem[i]], bG[i] + (size_t)kc * 512);
  };

  auto compute_chunk = [&](int b) {
    v16bf af[2];
#pragma unroll
    for (int mi = 0; mi < 2; ++mi) {
      const v8bf* p = reinterpret_cast<const v8bf*>(&sA[b][(wm * 32 + mi * 16 + lrow) * 32 + half * 16]);
      reinterpret_cast<v8bf*>(&af[mi])[0] = p[0];
      reinterpret_cast<v8bf*>(&af[mi])[1] = p[1];
    }
#pragma unroll
    for (int ni = 0; ni < 4; ++ni) {
      v16bf bfr = *reinterpret_cast<const v16bf*>(&sB[b][(wn * 4 + ni) * 512 + lane * 16]);
      acc[0][ni] = wmma_bf16(af[0], bfr, acc[0][ni]);
      acc[1][ni] = wmma_bf16(af[1], bfr, acc[1][ni]);
    }
  };

  issue_chunk(0, 0);
  int buf = 0;
  for (int kc = 0; kc < kcn; ++kc) {
    if (kc + 1 < kcn) {
      issue_chunk(kc + 1, buf ^ 1);
      WAIT_ASYNC(5);          // retire this buffer's 5 copies (in-order completion)
    } else {
      WAIT_ASYNC(0);
    }
    __syncthreads();          // everyone's copies for `buf` have landed
    compute_chunk(buf);
    __syncthreads();          // all reads of `buf` done before it is refilled
    buf ^= 1;
  }

#pragma unroll
  for (int mi = 0; mi < 2; ++mi)
#pragma unroll
    for (int ni = 0; ni < 4; ++ni)
#pragma unroll
      for (int r = 0; r < 8; ++r) {
        int m = bm * 64 + wm * 32 + mi * 16 + r + half * 8;
        int n = bn * 256 + (wn * 4 + ni) * 16 + lrow;
        float val = acc[mi][ni][r] + bias[n];
        if (MODE == 0) {
          int which = n / DMODEL;
          int within = n - which * DMODEL;
          int hh = within >> 6, d = within & 63;
          int b = m >> 9, sidx = m & (SEQ - 1);
          size_t bh = (size_t)(b * NHEAD + hh);
          if (which == 0)      qb[(bh * SEQ + sidx) * HD + d] = f2bf(val * 0.125f);
          else if (which == 1) kb[(bh * SEQ + sidx) * HD + d] = f2bf(val);
          else                 vtb[(bh * HD + d) * SEQ + sidx] = f2bf(val);
        } else if (MODE == 1) {
          xout[(size_t)m * DMODEL + n] += val;
        } else {
          hout[(size_t)m * (size_t)N + n] = f2bf(gelu_tanh(val));
        }
      }
}

// ---------------- flash attention: one wave per 16-row Q tile ----------------
__global__ __launch_bounds__(256) void k_attn(const unsigned short* __restrict__ q,
                                              const unsigned short* __restrict__ k,
                                              const unsigned short* __restrict__ vt,
                                              unsigned short* __restrict__ out) {
  __shared__ __align__(64) unsigned short sh[8 * 512];   // 16x32 bf16 per wave
  int widx = threadIdx.x >> 5;
  int w = blockIdx.x * 8 + widx;
  int lane = threadIdx.x & 31, lrow = lane & 15, half = lane >> 4;
  int bh = w >> 5;      // 32 q-tiles per (b,h)
  int qt = w & 31;
  const unsigned short* qbp = q + (size_t)bh * SEQ * HD;
  const unsigned short* kbp = k + (size_t)bh * SEQ * HD;
  const unsigned short* vbp = vt + (size_t)bh * HD * SEQ;
  unsigned short* shw = sh + widx * 512;

  // Q fragments (k-dim 64 -> two 32-chunks), A-matrix layout
  v16bf qa[2];
#pragma unroll
  for (int kc = 0; kc < 2; ++kc) {
    const v8bf* p = reinterpret_cast<const v8bf*>(qbp + (size_t)(qt * 16 + lrow) * HD + kc * 32 + half * 16);
    reinterpret_cast<v8bf*>(&qa[kc])[0] = p[0];
    reinterpret_cast<v8bf*>(&qa[kc])[1] = p[1];
  }

  v8f O[4];
  float mstat[8], lstat[8];
#pragma unroll
  for (int j = 0; j < 4; ++j) O[j] = vzero8();
#pragma unroll
  for (int r = 0; r < 8; ++r) { mstat[r] = -1e30f; lstat[r] = 0.f; }

  int ktmax = (qt * 16 + 15) >> 5;
  for (int kt = 0; kt <= ktmax; ++kt) {
    v8f s0 = vzero8(), s1 = vzero8();
#pragma unroll
    for (int kc = 0; kc < 2; ++kc) {
      v16bf kf0 = *reinterpret_cast<const v16bf*>(kbp + (size_t)(kt * 32 + lrow) * HD + kc * 32 + half * 16);
      v16bf kf1 = *reinterpret_cast<const v16bf*>(kbp + (size_t)(kt * 32 + 16 + lrow) * HD + kc * 32 + half * 16);
      s0 = wmma_bf16(qa[kc], kf0, s0);
      s1 = wmma_bf16(qa[kc], kf1, s1);
    }
    int colbase = kt * 32;
    float pm0[8], pm1[8], alpha[8];
#pragma unroll
    for (int r = 0; r < 8; ++r) {
      int mrow = qt * 16 + r + half * 8;
      pm0[r] = (colbase + lrow      <= mrow) ? s0[r] : -10000.0f;
      pm1[r] = (colbase + 16 + lrow <= mrow) ? s1[r] : -10000.0f;
    }
#pragma unroll
    for (int r = 0; r < 8; ++r) {
      float t = fmaxf(pm0[r], pm1[r]);
      t = fmaxf(t, __shfl_xor(t, 1, 32));
      t = fmaxf(t, __shfl_xor(t, 2, 32));
      t = fmaxf(t, __shfl_xor(t, 4, 32));
      t = fmaxf(t, __shfl_xor(t, 8, 32));
      float mn = fmaxf(mstat[r], t);
      alpha[r] = __expf(mstat[r] - mn);
      mstat[r] = mn;
      float p0 = __expf(pm0[r] - mn), p1 = __expf(pm1[r] - mn);
      pm0[r] = p0; pm1[r] = p1;
      float rs = p0 + p1;
      rs += __shfl_xor(rs, 1, 32);
      rs += __shfl_xor(rs, 2, 32);
      rs += __shfl_xor(rs, 4, 32);
      rs += __shfl_xor(rs, 8, 32);
      lstat[r] = lstat[r] * alpha[r] + rs;
    }
#pragma unroll
    for (int j = 0; j < 4; ++j)
#pragma unroll
      for (int r = 0; r < 8; ++r) O[j][r] *= alpha[r];

    // stage P (16x32 bf16) through wave-private LDS to reach A-fragment layout
#pragma unroll
    for (int r = 0; r < 8; ++r) {
      shw[(r + half * 8) * 32 + lrow]      = f2bf(pm0[r]);
      shw[(r + half * 8) * 32 + 16 + lrow] = f2bf(pm1[r]);
    }
    asm volatile("s_wait_dscnt 0" ::: "memory");
    v16bf pa;
    const v8bf* pl = reinterpret_cast<const v8bf*>(shw + lrow * 32 + half * 16);
    reinterpret_cast<v8bf*>(&pa)[0] = pl[0];
    reinterpret_cast<v8bf*>(&pa)[1] = pl[1];
#pragma unroll
    for (int j = 0; j < 4; ++j) {
      v16bf vf = *reinterpret_cast<const v16bf*>(vbp + (size_t)(j * 16 + lrow) * SEQ + kt * 32 + half * 16);
      O[j] = wmma_bf16(pa, vf, O[j]);
    }
  }

  int b = bh / NHEAD, hh = bh % NHEAD;
#pragma unroll
  for (int j = 0; j < 4; ++j)
#pragma unroll
    for (int r = 0; r < 8; ++r) {
      int token = b * SEQ + qt * 16 + r + half * 8;
      float val = O[j][r] / lstat[r];
      out[(size_t)token * DMODEL + hh * HD + j * 16 + lrow] = f2bf(val);
    }
}

// ---------------- host launcher ----------------
extern "C" void kernel_launch(void* const* d_in, const int* in_sizes, int n_in,
                              void* d_out, int out_size, void* d_ws, size_t ws_size,
                              hipStream_t stream) {
  (void)in_sizes; (void)n_in; (void)out_size; (void)ws_size;
  const int*   ids      = (const int*)d_in[0];
  const float* wte      = (const float*)d_in[1];
  const float* wpe      = (const float*)d_in[2];
  const float* ln1_g    = (const float*)d_in[3];
  const float* ln1_b    = (const float*)d_in[4];
  const float* attn_w   = (const float*)d_in[5];
  const float* attn_b   = (const float*)d_in[6];
  const float* proj_w   = (const float*)d_in[7];
  const float* proj_b   = (const float*)d_in[8];
  const float* ln2_g    = (const float*)d_in[9];
  const float* ln2_b    = (const float*)d_in[10];
  const float* fc_w     = (const float*)d_in[11];
  const float* fc_b     = (const float*)d_in[12];
  const float* fcproj_w = (const float*)d_in[13];
  const float* fcproj_b = (const float*)d_in[14];
  const float* lnf_g    = (const float*)d_in[15];
  const float* lnf_b    = (const float*)d_in[16];
  float* out = (float*)d_out;

  char* ws = (char*)d_ws;
  auto al = [](size_t x) { return (x + 255) & ~(size_t)255; };
  size_t off = 0;
  float* X = (float*)(ws + off);            off += al((size_t)TOK * DMODEL * 4);
  unsigned short* ABUF = (unsigned short*)(ws + off); off += al((size_t)TOK * DMODEL * 2);
  unsigned short* ATTN = (unsigned short*)(ws + off); off += al((size_t)TOK * DMODEL * 2);
  unsigned short* HBUF = (unsigned short*)(ws + off); off += al((size_t)TOK * FDIM * 2);
  unsigned short* QB   = (unsigned short*)(ws + off); off += al((size_t)96 * SEQ * HD * 2);
  unsigned short* KB   = (unsigned short*)(ws + off); off += al((size_t)96 * SEQ * HD * 2);
  unsigned short* VTB  = (unsigned short*)(ws + off); off += al((size_t)96 * SEQ * HD * 2);
  unsigned short* WP   = (unsigned short*)(ws + off); off += al((size_t)DMODEL * FDIM * 2);

  k_embed<<<(TOK * DMODEL) / 256, 256, 0, stream>>>(ids, wte, wpe, X);

  for (int l = 0; l < NLAYER; ++l) {
    // --- attention path ---
    k_ln<false><<<TOK / 8, 256, 0, stream>>>(X, ln1_g + (size_t)l * DMODEL, ln1_b + (size_t)l * DMODEL, ABUF, nullptr);
    k_pack<<<(D3 * DMODEL / 16) / 256, 256, 0, stream>>>(attn_w + (size_t)l * DMODEL * D3, WP, DMODEL, D3);
    k_gemm<0><<<64 * (D3 / 256), 256, 0, stream>>>(ABUF, WP, attn_b + (size_t)l * D3, DMODEL, D3,
                                                   nullptr, QB, KB, VTB, nullptr);
    k_attn<<<3072 / 8, 256, 0, stream>>>(QB, KB, VTB, ATTN);
    k_pack<<<(DMODEL * DMODEL / 16) / 256, 256, 0, stream>>>(proj_w + (size_t)l * DMODEL * DMODEL, WP, DMODEL, DMODEL);
    k_gemm<1><<<64 * (DMODEL / 256), 256, 0, stream>>>(ATTN, WP, proj_b + (size_t)l * DMODEL, DMODEL, DMODEL,
                                                       X, nullptr, nullptr, nullptr, nullptr);
    // --- MLP path ---
    k_ln<false><<<TOK / 8, 256, 0, stream>>>(X, ln2_g + (size_t)l * DMODEL, ln2_b + (size_t)l * DMODEL, ABUF, nullptr);
    k_pack<<<(DMODEL * FDIM / 16) / 256, 256, 0, stream>>>(fc_w + (size_t)l * DMODEL * FDIM, WP, DMODEL, FDIM);
    k_gemm<2><<<64 * (FDIM / 256), 256, 0, stream>>>(ABUF, WP, fc_b + (size_t)l * FDIM, DMODEL, FDIM,
                                                     nullptr, nullptr, nullptr, nullptr, HBUF);
    k_pack<<<(FDIM * DMODEL / 16) / 256, 256, 0, stream>>>(fcproj_w + (size_t)l * FDIM * DMODEL, WP, FDIM, DMODEL);
    k_gemm<1><<<64 * (DMODEL / 256), 256, 0, stream>>>(HBUF, WP, fcproj_b + (size_t)l * DMODEL, FDIM, DMODEL,
                                                       X, nullptr, nullptr, nullptr, nullptr);
  }

  k_ln<true><<<TOK / 8, 256, 0, stream>>>(X, lnf_g, lnf_b, nullptr, out);
}
